// DS2_18356690223700
// MI455X (gfx1250) — matrix operand order repaired and checked by reading, not compiled, not run on hardware
//
#include <hip/hip_runtime.h>
#include <math.h>

// ---------------------------------------------------------------------------
// CDNA5 (gfx1250) wave32 WMMA types & helpers
// ---------------------------------------------------------------------------
typedef __attribute__((ext_vector_type(16))) _Float16 v16h;
typedef __attribute__((ext_vector_type(8)))  float    v8f;
typedef __attribute__((ext_vector_type(4)))  unsigned u32x4;   // POD 128-bit

#define DEV static __device__ __forceinline__

union U16x8 { u32x4 q[2]; v16h v; };

DEV v8f wmma16(v16h a, v16h b, v8f c) {
  // D = A(16x32 f16) * B(32x16 f16) + C(16x16 f32)
  return __builtin_amdgcn_wmma_f32_16x16x32_f16(
      /*neg_a=*/false, a, /*neg_b=*/false, b,
      /*c_mod=*/(short)0, c, /*reuse_a=*/false, /*reuse_b=*/false);
}

// A fragment: row-major A[M][K] f16; lane l -> m = l&15, hi = l>>4.
// ISA 16-bit A 16x32 layout: VGPR v, half t: k = ((v&4)<<2) + hi*8 + (v&3)*2 + t.
// -> dwords (m*K+k0)/2 + hi*4 + {0..3} and +8+{0..3}: two aligned b128 loads.
DEV v16h load_a16(const _Float16* A, int m, int hi, int ldk, int k0) {
  U16x8 f;
  const u32x4* p = (const u32x4*)(A + (size_t)m * (size_t)ldk + (size_t)k0);
  f.q[0] = p[hi];
  f.q[1] = p[hi + 2];
  return f.v;
}

// B fragment from pre-transposed, N-padded Bt[Npad][K] f16; lane l -> n, hi.
// B 32x16 layout (same k order as A): VGPR v, half t:
// k = ((v&4)<<2) + hi*8 + (v&3)*2 + t  -> two aligned b128 loads.
DEV v16h load_b16(const _Float16* Bt, int n, int hi, int ldk, int k0) {
  U16x8 f;
  const u32x4* p =
      (const u32x4*)(Bt + (size_t)n * (size_t)ldk + (size_t)k0);
  f.q[0] = p[hi];
  f.q[1] = p[hi + 2];
  return f.v;
}

// ---------------------------------------------------------------------------
// Generic WMMA GEMM: C[M,N] = A[M,K]*Bt^T + epilogue.
// Block = 128 threads = 4 waves; wave computes a 16M x 64N tile.
// grid = (ceil(N/256), M/16).  M multiple of 16; Bt has ceil(N/256)*256 rows.
// ---------------------------------------------------------------------------
enum { EP_BIAS = 1, EP_RELU = 2, EP_CLIP = 4, EP_AFF = 8, EP_F16 = 16 };

template <int EPI>
__global__ __launch_bounds__(128)
void gemm_f16(const _Float16* __restrict__ A, const _Float16* __restrict__ Bt,
              const float* __restrict__ bias, const float* __restrict__ scale,
              const float* __restrict__ shift, void* __restrict__ Cout,
              int N, int K, float clipv) {
  const int lane = threadIdx.x & 31;
  const int wv   = threadIdx.x >> 5;
  const int m0   = blockIdx.y << 4;
  const int n0   = (blockIdx.x << 8) + (wv << 6);
  if (n0 >= N) return;                       // wave-uniform: EXEC stays all-ones
  const int hi = lane >> 4, nl = lane & 15;

  const v8f zero = {0.f, 0.f, 0.f, 0.f, 0.f, 0.f, 0.f, 0.f};
  v8f acc0 = zero, acc1 = zero, acc2 = zero, acc3 = zero;

  for (int k0 = 0; k0 < K; k0 += 32) {
    v16h a  = load_a16(A, m0 + nl, hi, K, k0);
    v16h b0 = load_b16(Bt, n0 + nl,      hi, K, k0);
    v16h b1 = load_b16(Bt, n0 + 16 + nl, hi, K, k0);
    v16h b2 = load_b16(Bt, n0 + 32 + nl, hi, K, k0);
    v16h b3 = load_b16(Bt, n0 + 48 + nl, hi, K, k0);
    acc0 = wmma16(a, b0, acc0);
    acc1 = wmma16(a, b1, acc1);
    acc2 = wmma16(a, b2, acc2);
    acc3 = wmma16(a, b3, acc3);
  }

  // C/D layout: VGPR v, lanes 0-15 -> (M=v, N=lane); lanes 16-31 -> (M=8+v).
  const v8f acc[4] = {acc0, acc1, acc2, acc3};
#pragma unroll
  for (int t = 0; t < 4; ++t) {
    int nn = n0 + t * 16 + nl;
    if (nn >= N) continue;
    float bs = (EPI & EP_BIAS) ? bias[nn] : 0.f;
    float sc = (EPI & EP_AFF) ? scale[nn] : 1.f;
    float sh = (EPI & EP_AFF) ? shift[nn] : 0.f;
#pragma unroll
    for (int v = 0; v < 8; ++v) {
      int mm = m0 + v + (hi << 3);
      float x = acc[t][v] + bs;
      if (EPI & EP_RELU) x = fmaxf(x, 0.f);
      if (EPI & EP_CLIP) x = fminf(x, clipv);
      if (EPI & EP_AFF)  x = x * sc + sh;
      size_t off = (size_t)mm * (size_t)N + (size_t)nn;
      if (EPI & EP_F16) ((_Float16*)Cout)[off] = (_Float16)x;
      else              ((float*)Cout)[off] = x;
    }
  }
}

// ---------------------------------------------------------------------------
// GRU recurrence (both directions in one launch; gridDim.y = dir).
// 8 blocks per dir, 3 waves/block (one per gate z/r/h), each wave: 16x64 WMMA
// over K=512.  h is double-buffered in global; per-step grid barrier per dir.
// ---------------------------------------------------------------------------
DEV void dirbar(unsigned* c, unsigned phase, unsigned nb) {
  __syncthreads();
  if (threadIdx.x == 0) {
    __threadfence();
    atomicAdd(c, 1u);
    volatile unsigned* vc = (volatile unsigned*)c;
    while (*vc < phase * nb) { __builtin_amdgcn_s_sleep(1); }
    __threadfence();
  }
  __syncthreads();
}

__global__ __launch_bounds__(96)
void gru_rec(const float* __restrict__ xp,       // [2][16*T][1536]
             const _Float16* __restrict__ WhtL,  // [2][1536][512] transposed f16
             const float* __restrict__ gbL,      // gru_b + l*4*1536
             float* __restrict__ hF,             // [2][2][16][512]
             _Float16* __restrict__ h16,         // [2][2][16][512]
             float* __restrict__ y,              // [2][16*T][512]
             unsigned* __restrict__ ctr,         // [2]
             int T) {
  const int H = 512, G3 = 1536, NB = 8;
  const int d    = blockIdx.y;
  const int j0   = blockIdx.x << 6;      // 64-column strip of h
  const int gate = threadIdx.x >> 5;     // 0=z, 1=r, 2=h
  const int lane = threadIdx.x & 31;
  const int hi = lane >> 4, nl = lane & 15;

  const float*    xpd  = xp   + (size_t)d * 16u * (size_t)T * G3;
  const _Float16* Wt   = WhtL + (size_t)d * G3 * H;
  const float*    bh   = gbL  + (d * 2 + 1) * G3;
  float*          hFd  = hF   + (size_t)d * 2 * 16 * H;
  _Float16*       h16d = h16  + (size_t)d * 2 * 16 * H;
  float*          yd   = y    + (size_t)d * 16u * (size_t)T * H;
  unsigned*       c    = ctr + d;

  __shared__ float lds[3][16][64];

  // zero initial hidden state (phase buffer 0), this block's strip
  for (int idx = threadIdx.x; idx < 16 * 64; idx += 96) {
    int m = idx >> 6, j = j0 + (idx & 63);
    hFd[m * H + j]  = 0.f;
    h16d[m * H + j] = (_Float16)0.f;
  }
  unsigned phase = 0;
  dirbar(c, ++phase, NB);

  const int cb = gate * H + j0;          // column base inside 3H

  for (int s = 0; s < T; ++s) {
    const int t = d ? (T - 1 - s) : s;
    const int cur = s & 1, nxt = cur ^ 1;
    const _Float16* hA = h16d + cur * 16 * H;

    const v8f zero = {0.f, 0.f, 0.f, 0.f, 0.f, 0.f, 0.f, 0.f};
    v8f acc0 = zero, acc1 = zero, acc2 = zero, acc3 = zero;
    for (int k0 = 0; k0 < H; k0 += 32) {
      v16h a  = load_a16(hA, nl, hi, H, k0);
      v16h b0 = load_b16(Wt, cb + nl,      hi, H, k0);
      v16h b1 = load_b16(Wt, cb + 16 + nl, hi, H, k0);
      v16h b2 = load_b16(Wt, cb + 32 + nl, hi, H, k0);
      v16h b3 = load_b16(Wt, cb + 48 + nl, hi, H, k0);
      acc0 = wmma16(a, b0, acc0);
      acc1 = wmma16(a, b1, acc1);
      acc2 = wmma16(a, b2, acc2);
      acc3 = wmma16(a, b3, acc3);
    }
    // deposit m-h tile (+recurrent bias) into LDS
    const v8f acc[4] = {acc0, acc1, acc2, acc3};
#pragma unroll
    for (int tt = 0; tt < 4; ++tt) {
      int jj = tt * 16 + nl;
      float bb = bh[cb + jj];
#pragma unroll
      for (int v = 0; v < 8; ++v)
        lds[gate][v + (hi << 3)][jj] = acc[tt][v] + bb;
    }
    __syncthreads();

    // elementwise gate combine for this 64-column strip
    for (int idx = threadIdx.x; idx < 16 * 64; idx += 96) {
      int m = idx >> 6, j = idx & 63;
      int col = j0 + j;
      size_t xb = ((size_t)m * (size_t)T + (size_t)t) * G3 + (size_t)col;
      float xz = xpd[xb], xr = xpd[xb + H], xh = xpd[xb + 2 * H];
      float z  = 1.f / (1.f + __expf(-(xz + lds[0][m][j])));
      float r  = 1.f / (1.f + __expf(-(xr + lds[1][m][j])));
      float hh = fmaxf(xh + r * lds[2][m][j], 0.f);
      float ho = hFd[cur * 16 * H + m * H + col];
      float hn = z * ho + (1.f - z) * hh;
      hFd[nxt * 16 * H + m * H + col]  = hn;
      h16d[nxt * 16 * H + m * H + col] = (_Float16)hn;
      yd[((size_t)m * (size_t)T + (size_t)t) * H + col] = hn;
    }
    dirbar(c, ++phase, NB);   // also protects LDS reuse next iteration
  }
}

// ---------------------------------------------------------------------------
// Elementwise / packing kernels
// ---------------------------------------------------------------------------
__global__ void k_bnprep(const float* g, const float* b, const float* m,
                         const float* v, float* scale, float* shift, int n) {
  int i = blockIdx.x * 256 + threadIdx.x;
  if (i < n) {
    float s = g[i] * rsqrtf(v[i] + 1e-3f);
    scale[i] = s;
    shift[i] = b[i] - m[i] * s;
  }
}

__global__ void k_xpre(const float* x, const float* scale, const float* shift,
                       _Float16* out, int total, int F) {
  int i = blockIdx.x * 256 + threadIdx.x;
  if (i < total) {
    int f = i % F;
    out[i] = (_Float16)(x[i] * scale[f] + shift[f]);
  }
}

// in[K][N] f32 (row-major) -> out[Npad][Kp] f16, zero-padded in both dims
__global__ void k_packwt(const float* in, _Float16* out, int K, int Kp, int N,
                         int total) {
  int i = blockIdx.x * 256 + threadIdx.x;
  if (i < total) {
    int n = i / Kp, k = i % Kp;
    out[i] = (k < K && n < N) ? (_Float16)in[(size_t)k * (size_t)N + n]
                              : (_Float16)0.f;
  }
}

// x[16][1600][80] f16 -> col[12768][416] f16 (K=5, stride 2, pad 400->416)
__global__ void k_im2col1(const _Float16* x, _Float16* col, int total) {
  int i = blockIdx.x * 256 + threadIdx.x;
  if (i < total) {
    int row = i / 416, k = i % 416;
    _Float16 v = (_Float16)0.f;
    if (k < 400) {
      int kk = k / 80, f = k % 80;
      int b = row / 798, ot = row % 798;
      v = x[((size_t)b * 1600 + (size_t)(ot * 2 + kk)) * 80 + f];
    }
    col[i] = v;
  }
}

// x[16][798][512] f16 -> col[6352][2560] f16 (K=5, stride 2)
__global__ void k_im2col2(const _Float16* x, _Float16* col, int total) {
  int i = blockIdx.x * 256 + threadIdx.x;
  if (i < total) {
    int row = i / 2560, k = i % 2560;
    int kk = k / 512, c2 = k % 512;
    int b = row / 397, ot = row % 397;
    col[i] = x[((size_t)b * 798 + (size_t)(ot * 2 + kk)) * 512 + c2];
  }
}

template <bool AFF>
__global__ void k_combine(const float* y0, const float* y1, const float* scale,
                          const float* shift, _Float16* out, int total, int H) {
  int i = blockIdx.x * 256 + threadIdx.x;
  if (i < total) {
    float v = y0[i] + y1[i];
    if (AFF) { int c = i % H; v = v * scale[c] + shift[c]; }
    out[i] = (_Float16)v;
  }
}

__global__ __launch_bounds__(256)
void k_softmax(float* out, int cols) {
  __shared__ float red[256];
  float* p = out + (size_t)blockIdx.x * (size_t)cols;
  float mx = -3.4e38f;
  for (int c = threadIdx.x; c < cols; c += 256) mx = fmaxf(mx, p[c]);
  red[threadIdx.x] = mx; __syncthreads();
  for (int s = 128; s > 0; s >>= 1) {
    if (threadIdx.x < s) red[threadIdx.x] = fmaxf(red[threadIdx.x], red[threadIdx.x + s]);
    __syncthreads();
  }
  mx = red[0]; __syncthreads();
  float sum = 0.f;
  for (int c = threadIdx.x; c < cols; c += 256) {
    float e = __expf(p[c] - mx); p[c] = e; sum += e;
  }
  red[threadIdx.x] = sum; __syncthreads();
  for (int s = 128; s > 0; s >>= 1) {
    if (threadIdx.x < s) red[threadIdx.x] += red[threadIdx.x + s];
    __syncthreads();
  }
  float inv = 1.f / red[0];
  for (int c = threadIdx.x; c < cols; c += 256) p[c] *= inv;
}

// ---------------------------------------------------------------------------
// Host orchestration
// ---------------------------------------------------------------------------
extern "C" void kernel_launch(void* const* d_in, const int* in_sizes, int n_in,
                              void* d_out, int out_size, void* d_ws, size_t ws_size,
                              hipStream_t stream) {
  const float* x       = (const float*)d_in[0];
  const float* bn1_g   = (const float*)d_in[1];
  const float* bn1_b   = (const float*)d_in[2];
  const float* bn1_m   = (const float*)d_in[3];
  const float* bn1_v   = (const float*)d_in[4];
  const float* conv_w0 = (const float*)d_in[5];
  const float* conv_b0 = (const float*)d_in[6];
  const float* conv_w1 = (const float*)d_in[7];
  const float* conv_b1 = (const float*)d_in[8];
  const float* bn2_g   = (const float*)d_in[9];
  const float* bn2_b   = (const float*)d_in[10];
  const float* bn2_m   = (const float*)d_in[11];
  const float* bn2_v   = (const float*)d_in[12];
  const float* gru_wx  = (const float*)d_in[13];
  const float* gru_wh  = (const float*)d_in[14];
  const float* gru_b   = (const float*)d_in[15];
  const float* bn3_g   = (const float*)d_in[16];
  const float* bn3_b   = (const float*)d_in[17];
  const float* bn3_m   = (const float*)d_in[18];
  const float* bn3_v   = (const float*)d_in[19];
  const float* fc_w    = (const float*)d_in[20];
  const float* fc_b    = (const float*)d_in[21];
  const float* sm_w    = (const float*)d_in[22];
  const float* sm_b    = (const float*)d_in[23];

  const int B = 16, T0 = 1600, F = 80, C = 512, T1 = 798, T2 = 397;
  const int G3 = 1536, V = 6000, Vp = 6144;     // Vp: N padded to 256 multiple
  const int M1 = B * T1;          // 12768 (mult of 16)
  const int M2 = B * T2;          // 6352  (mult of 16)
  const int K1 = 416, K2 = 2560;  // both mult of 32 (K1 padded from 400)

  char* w = (char*)d_ws;
  auto alloc = [&](size_t bytes) -> void* {
    void* p = (void*)w;
    w += (bytes + 255) & ~(size_t)255;
    return p;
  };

  float* bn1s = (float*)alloc(F * 4);
  float* bn1h = (float*)alloc(F * 4);
  float* bn2s = (float*)alloc(C * 4);
  float* bn2h = (float*)alloc(C * 4);
  float* bn3s = (float*)alloc(C * 4);
  float* bn3h = (float*)alloc(C * 4);
  _Float16* xbn  = (_Float16*)alloc((size_t)B * T0 * F * 2);
  _Float16* col1 = (_Float16*)alloc((size_t)M1 * K1 * 2);
  _Float16* c1o  = (_Float16*)alloc((size_t)M1 * C * 2);
  _Float16* col2 = (_Float16*)alloc((size_t)M2 * K2 * 2);
  _Float16* xin  = (_Float16*)alloc((size_t)M2 * C * 2);
  float*    xp   = (float*)alloc((size_t)2 * M2 * G3 * 4);
  float*    hF   = (float*)alloc((size_t)2 * 2 * 16 * C * 4);
  _Float16* h16  = (_Float16*)alloc((size_t)2 * 2 * 16 * C * 2);
  float*    ybuf = (float*)alloc((size_t)2 * M2 * C * 4);
  _Float16* fco  = (_Float16*)alloc((size_t)M2 * C * 2);
  _Float16* w0t  = (_Float16*)alloc((size_t)C * K1 * 2);
  _Float16* w1t  = (_Float16*)alloc((size_t)C * K2 * 2);
  _Float16* wxt  = (_Float16*)alloc((size_t)6 * G3 * C * 2);
  _Float16* wht  = (_Float16*)alloc((size_t)6 * G3 * C * 2);
  _Float16* fct  = (_Float16*)alloc((size_t)C * C * 2);
  _Float16* smt  = (_Float16*)alloc((size_t)Vp * C * 2);   // padded rows
  unsigned* ctr  = (unsigned*)alloc(6 * sizeof(unsigned));

  hipMemsetAsync(ctr, 0, 6 * sizeof(unsigned), stream);

  // BN affine prep
  k_bnprep<<<1, 256, 0, stream>>>(bn1_g, bn1_b, bn1_m, bn1_v, bn1s, bn1h, F);
  k_bnprep<<<2, 256, 0, stream>>>(bn2_g, bn2_b, bn2_m, bn2_v, bn2s, bn2h, C);
  k_bnprep<<<2, 256, 0, stream>>>(bn3_g, bn3_b, bn3_m, bn3_v, bn3s, bn3h, C);

  // Weight pack (f32 [K][N] -> f16 [Npad][Kp], zero-padded both dims)
  auto packw = [&](const float* src, _Float16* dst, int K, int Kp, int N,
                   int Npad) {
    int total = Npad * Kp;
    k_packwt<<<(total + 255) / 256, 256, 0, stream>>>(src, dst, K, Kp, N, total);
  };
  packw(conv_w0, w0t, 400, K1, C, C);
  packw(conv_w1, w1t, K2, K2, C, C);
  for (int l = 0; l < 3; ++l)
    for (int d = 0; d < 2; ++d) {
      size_t o = (size_t)(l * 2 + d);
      packw(gru_wx + o * C * G3, wxt + o * G3 * C, C, C, G3, G3);
      packw(gru_wh + o * C * G3, wht + o * G3 * C, C, C, G3, G3);
    }
  packw(fc_w, fct, C, C, C, C);
  packw(sm_w, smt, C, C, V, Vp);

  // BN1 + f16 convert
  {
    int tx = B * T0 * F;
    k_xpre<<<(tx + 255) / 256, 256, 0, stream>>>(x, bn1s, bn1h, xbn, tx, F);
  }

  // conv1: im2col + WMMA GEMM (+bias, relu)
  {
    int t1 = M1 * K1;
    k_im2col1<<<(t1 + 255) / 256, 256, 0, stream>>>(xbn, col1, t1);
    gemm_f16<EP_BIAS | EP_RELU | EP_F16>
        <<<dim3(2, M1 / 16), 128, 0, stream>>>(col1, w0t, conv_b0, nullptr,
                                               nullptr, c1o, C, K1, 0.f);
  }
  // conv2: im2col + WMMA GEMM (+bias, relu, BN2 affine)
  {
    int t2 = M2 * K2;
    k_im2col2<<<(t2 + 255) / 256, 256, 0, stream>>>(c1o, col2, t2);
    gemm_f16<EP_BIAS | EP_RELU | EP_AFF | EP_F16>
        <<<dim3(2, M2 / 16), 128, 0, stream>>>(col2, w1t, conv_b1, bn2s, bn2h,
                                               xin, C, K2, 0.f);
  }

  // 3 bi-GRU layers (sum of directions)
  for (int l = 0; l < 3; ++l) {
    for (int d = 0; d < 2; ++d) {
      size_t o = (size_t)(l * 2 + d);
      gemm_f16<EP_BIAS><<<dim3(6, M2 / 16), 128, 0, stream>>>(
          xin, wxt + o * G3 * C, gru_b + (o * 2 + 0) * G3, nullptr, nullptr,
          xp + (size_t)d * M2 * G3, G3, C, 0.f);
    }
    gru_rec<<<dim3(8, 2), 96, 0, stream>>>(
        xp, wht + (size_t)l * 2 * G3 * C, gru_b + (size_t)l * 4 * G3,
        hF, h16, ybuf, ctr + l * 2, T2);
    int tc = M2 * C;
    if (l == 2)
      k_combine<true><<<(tc + 255) / 256, 256, 0, stream>>>(
          ybuf, ybuf + (size_t)M2 * C, bn3s, bn3h, xin, tc, C);
    else
      k_combine<false><<<(tc + 255) / 256, 256, 0, stream>>>(
          ybuf, ybuf + (size_t)M2 * C, nullptr, nullptr, xin, tc, C);
  }

  // FC (bias + relu + clip 20) -> f16
  gemm_f16<EP_BIAS | EP_RELU | EP_CLIP | EP_F16>
      <<<dim3(2, M2 / 16), 128, 0, stream>>>(xin, fct, fc_b, nullptr, nullptr,
                                             fco, C, C, 20.f);
  // softmax matmul -> logits f32 in d_out, then row softmax in-place
  gemm_f16<EP_BIAS><<<dim3((V + 255) / 256, M2 / 16), 128, 0, stream>>>(
      fco, smt, sm_b, nullptr, nullptr, (float*)d_out, V, C, 0.f);
  k_softmax<<<M2, 256, 0, stream>>>((float*)d_out, V);
}
